// GATLayer_87548613362381
// MI455X (gfx1250) — matrix operand order repaired and checked by reading, not compiled, not run on hardware
//
#include <hip/hip_runtime.h>
#include <math.h>

#define NNODES 8192
#define DIM 256
#define NWORDS (NNODES / 32)
#define NCHUNK (NNODES / 32)
#define NEG_SLOPE 0.2f

typedef __attribute__((ext_vector_type(16))) __bf16 bf16x16;
typedef __attribute__((ext_vector_type(8)))  float  f32x8;

union BfTile {
    bf16x16 v;
    uint4 q[2];
    unsigned short h[16];
};

__device__ __forceinline__ unsigned short f2bf(float f) {
    unsigned u = __builtin_bit_cast(unsigned, f);
    u += 0x7fffu + ((u >> 16) & 1u);
    return (unsigned short)(u >> 16);
}

// generic pointer -> raw 32-bit LDS byte address (for async-to-LDS asm)
__device__ __forceinline__ unsigned lds_addr_of(const void* p) {
    return (unsigned)(unsigned long long)(__attribute__((address_space(3))) const void*)p;
}

// issue two async b128 copies: 32B global -> 32B LDS (same 16B delta on both
// sides, so instruction offset covers the second half)
__device__ __forceinline__ void stage_async32(unsigned lds, unsigned long long gaddr) {
    asm volatile("global_load_async_to_lds_b128 %0, %1, off"
                 :: "v"(lds), "v"(gaddr) : "memory");
    asm volatile("global_load_async_to_lds_b128 %0, %1, off offset:16"
                 :: "v"(lds), "v"(gaddr) : "memory");
}

__device__ __forceinline__ void wait_async0() {
    asm volatile("s_wait_asynccnt 0" ::: "memory");
}

// ---------------------------------------------------------------------------
// Transpose-convert W (DIM x DIM f32, row-major in->out) to WT (out x in, bf16)
// ---------------------------------------------------------------------------
__global__ void k_wt(const float* __restrict__ W, unsigned short* __restrict__ WT) {
    int k = threadIdx.x;      // fan-in index
    int n = blockIdx.x;       // fan-out index
    WT[n * DIM + k] = f2bf(W[k * DIM + n]);
}

// ---------------------------------------------------------------------------
// Pack adjacency (int32 0/1) into bitmask words: one wave -> one u32 word
// ---------------------------------------------------------------------------
__global__ void k_pack(const int* __restrict__ adj, unsigned int* __restrict__ packed) {
    size_t gid = (size_t)blockIdx.x * 256 + threadIdx.x;
    int a = adj[gid];
    unsigned long long b = __ballot(a != 0);
    if ((threadIdx.x & 31) == 0) packed[gid >> 5] = (unsigned int)b;
}

// ---------------------------------------------------------------------------
// Wh = X @ W via WMMA; writes WhT (bf16, DIM x NNODES transposed) and
// src[i] = Wh[i,:]@a_src, dst[i] = Wh[i,:]@a_dst.
// One wave per 16-row block; 16 accumulator tiles cover all 256 columns.
// ---------------------------------------------------------------------------
__global__ void __launch_bounds__(256)
k_gemm(const float* __restrict__ X, const unsigned short* __restrict__ WT,
       const float* __restrict__ a_src, const float* __restrict__ a_dst,
       unsigned short* __restrict__ WhT,
       float* __restrict__ srcv, float* __restrict__ dstv) {
    const int lane = threadIdx.x & 31;
    const int wave = threadIdx.x >> 5;
    const int l16  = lane & 15;
    const int hi   = lane >> 4;           // which 16-lane half
    const int kb   = hi * 8;              // A-operand K interleave offset
    const int i0   = (blockIdx.x * 8 + wave) * 16;
    const int irow = i0 + l16;

    f32x8 acc[16] = {};

    for (int k0 = 0; k0 < DIM; k0 += 32) {
        // A operand: X row irow, f32 -> bf16.  elems 0..7 <-> K=k0+kb+e,
        // elems 8..15 <-> K=k0+16+kb+(e-8)
        const float* xp = X + (size_t)irow * DIM + k0 + kb;
        float4 xa = *(const float4*)(xp);
        float4 xb = *(const float4*)(xp + 4);
        float4 xc = *(const float4*)(xp + 16);
        float4 xd = *(const float4*)(xp + 20);
        BfTile A;
        A.h[0]  = f2bf(xa.x); A.h[1]  = f2bf(xa.y); A.h[2]  = f2bf(xa.z); A.h[3]  = f2bf(xa.w);
        A.h[4]  = f2bf(xb.x); A.h[5]  = f2bf(xb.y); A.h[6]  = f2bf(xb.z); A.h[7]  = f2bf(xb.w);
        A.h[8]  = f2bf(xc.x); A.h[9]  = f2bf(xc.y); A.h[10] = f2bf(xc.z); A.h[11] = f2bf(xc.w);
        A.h[12] = f2bf(xd.x); A.h[13] = f2bf(xd.y); A.h[14] = f2bf(xd.z); A.h[15] = f2bf(xd.w);

        #pragma unroll
        for (int t = 0; t < 16; ++t) {
            // B operand: column c = 16t + l16, K = k0+kb+(0..7), k0+16+kb+(0..7)
            const unsigned short* wp = WT + (size_t)(t * 16 + l16) * DIM + k0 + kb;
            BfTile B;
            B.q[0] = *(const uint4*)(wp);
            B.q[1] = *(const uint4*)(wp + 16);
            acc[t] = __builtin_amdgcn_wmma_f32_16x16x32_bf16(
                false, A.v, false, B.v, (short)0, acc[t], false, false);
        }
    }

    // per-row attention logits: reduce over features
    float psrc[8] = {0.f,0.f,0.f,0.f,0.f,0.f,0.f,0.f};
    float pdst[8] = {0.f,0.f,0.f,0.f,0.f,0.f,0.f,0.f};
    #pragma unroll
    for (int t = 0; t < 16; ++t) {
        float fs = a_src[t * 16 + l16];
        float fd = a_dst[t * 16 + l16];
        #pragma unroll
        for (int r = 0; r < 8; ++r) {
            psrc[r] += acc[t][r] * fs;
            pdst[r] += acc[t][r] * fd;
        }
    }
    #pragma unroll
    for (int off = 1; off < 16; off <<= 1) {
        #pragma unroll
        for (int r = 0; r < 8; ++r) {
            psrc[r] += __shfl_xor(psrc[r], off, 32);
            pdst[r] += __shfl_xor(pdst[r], off, 32);
        }
    }
    if (l16 == 0) {
        #pragma unroll
        for (int r = 0; r < 8; ++r) {
            srcv[i0 + hi * 8 + r] = psrc[r];
            dstv[i0 + hi * 8 + r] = pdst[r];
        }
    }

    // write WhT (transposed, bf16): lane stores 8 consecutive node-indices per tile
    #pragma unroll
    for (int t = 0; t < 16; ++t) {
        BfTile O;
        #pragma unroll
        for (int r = 0; r < 8; ++r) O.h[r] = f2bf(acc[t][r]);
        *(uint4*)(WhT + (size_t)(t * 16 + l16) * NNODES + i0 + hi * 8) = O.q[0];
    }
}

// ---------------------------------------------------------------------------
// Per-row softmax stats over masked e = leaky_relu(src_i + dst_j).
// One wave per row; two passes over the (L2-hot) packed bitmask.
// ---------------------------------------------------------------------------
__global__ void __launch_bounds__(256)
k_stats(const unsigned int* __restrict__ packed,
        const float* __restrict__ srcv, const float* __restrict__ dstv,
        float* __restrict__ mrow, float* __restrict__ srow) {
    const int lane = threadIdx.x & 31;
    const int wave = threadIdx.x >> 5;
    const int row  = blockIdx.x * 8 + wave;
    const float sc = srcv[row];
    const unsigned int* prow = packed + (size_t)row * NWORDS;

    float m = -INFINITY;
    for (int p = 0; p < 8; ++p) {
        int w = p * 32 + lane;
        unsigned int bits = prow[w];
        if (bits) {
            const float* dp = dstv + w * 32;
            #pragma unroll
            for (int q4 = 0; q4 < 8; ++q4) {
                float4 d = *(const float4*)(dp + q4 * 4);
                unsigned int bb = bits >> (q4 * 4);
                float e;
                e = sc + d.x; e = e > 0.f ? e : NEG_SLOPE * e; if (bb & 1u) m = fmaxf(m, e);
                e = sc + d.y; e = e > 0.f ? e : NEG_SLOPE * e; if (bb & 2u) m = fmaxf(m, e);
                e = sc + d.z; e = e > 0.f ? e : NEG_SLOPE * e; if (bb & 4u) m = fmaxf(m, e);
                e = sc + d.w; e = e > 0.f ? e : NEG_SLOPE * e; if (bb & 8u) m = fmaxf(m, e);
            }
        }
    }
    for (int off = 1; off < 32; off <<= 1) m = fmaxf(m, __shfl_xor(m, off, 32));

    float s = 0.f;
    for (int p = 0; p < 8; ++p) {
        int w = p * 32 + lane;
        unsigned int bits = prow[w];
        if (bits) {
            const float* dp = dstv + w * 32;
            #pragma unroll
            for (int q4 = 0; q4 < 8; ++q4) {
                float4 d = *(const float4*)(dp + q4 * 4);
                unsigned int bb = bits >> (q4 * 4);
                float e;
                e = sc + d.x; e = e > 0.f ? e : NEG_SLOPE * e; s += (bb & 1u) ? __expf(e - m) : 0.f;
                e = sc + d.y; e = e > 0.f ? e : NEG_SLOPE * e; s += (bb & 2u) ? __expf(e - m) : 0.f;
                e = sc + d.z; e = e > 0.f ? e : NEG_SLOPE * e; s += (bb & 4u) ? __expf(e - m) : 0.f;
                e = sc + d.w; e = e > 0.f ? e : NEG_SLOPE * e; s += (bb & 8u) ? __expf(e - m) : 0.f;
            }
        }
    }
    for (int off = 1; off < 32; off <<= 1) s += __shfl_xor(s, off, 32);

    if (lane == 0) { mrow[row] = m; srow[row] = s; }
}

// ---------------------------------------------------------------------------
// Fused attention: H^T tile = WhT tile @ alpha^T tile over all neighbors j.
// Wave owns 16 nodes x 128 features (8 WMMA accumulators). A-operand (WhT
// slice) is staged into double-buffered LDS with GLOBAL_LOAD_ASYNC_TO_LDS_B128
// (ASYNCcnt), so the DMA for chunk c+1 overlaps the WMMA/exp work of chunk c.
// LDS pitch 40 ushorts = 80B -> conflict-free ds_load_b128 for A tiles.
// ---------------------------------------------------------------------------
__global__ void __launch_bounds__(256)
k_attn(const unsigned short* __restrict__ WhT, const unsigned int* __restrict__ packed,
       const float* __restrict__ srcv, const float* __restrict__ dstv,
       const float* __restrict__ mrow, const float* __restrict__ srow,
       float* __restrict__ Hout) {
    __shared__ __align__(16) unsigned short tileA[2][128 * 40];

    const int tid  = threadIdx.x;
    const int lane = tid & 31;
    const int wave = tid >> 5;
    const int l16  = lane & 15;
    const int hi   = lane >> 4;
    const int kb   = hi * 8;                      // unused: A and B use contiguous k = 16*hi + e
    const int i0   = (blockIdx.x * 8 + wave) * 16;
    const int irow = i0 + l16;                    // this lane's node
    const int nbase = blockIdx.y * 128;           // feature half

    const float sc = srcv[irow];
    const float mx = mrow[irow];
    const unsigned int* prow = packed + (size_t)irow * NWORDS;

    f32x8 acc[8] = {};

    // staging: 256 threads cover 128 rows x 2 half-chunks of 32B each
    const int srn   = tid >> 1;
    const int shalf = tid & 1;
    unsigned long long ga =
        (unsigned long long)(WhT + (size_t)(nbase + srn) * NNODES + shalf * 16);
    const unsigned lb0 = lds_addr_of(&tileA[0][srn * 40 + shalf * 16]);
    const unsigned lb1 = lds_addr_of(&tileA[1][srn * 40 + shalf * 16]);

    // prologue: async-stage chunk 0 into buffer 0
    stage_async32(lb0, ga);

    for (int c = 0; c < NCHUNK; ++c) {
        const int j0 = c * 32;
        wait_async0();                       // this wave's staged slice is in LDS
        __syncthreads();                     // everyone's slice is in LDS; prior
                                             // reads of the other buffer are done
        if (c + 1 < NCHUNK) {               // overlap: DMA next chunk now
            stage_async32((c & 1) ? lb0 : lb1, ga + 64);
        }
        ga += 64;
        const unsigned short* buf = tileA[c & 1];

        // B operand: lane holds p for node irow, j = j0 + 16*hi + (0..15)
        unsigned int bits = prow[j0 >> 5];
        const float* dp = dstv + j0 + hi * 16;
        float4 d0 = *(const float4*)(dp);
        float4 d1 = *(const float4*)(dp + 4);
        float4 d2 = *(const float4*)(dp + 8);
        float4 d3 = *(const float4*)(dp + 12);
        float dd[16] = {d0.x,d0.y,d0.z,d0.w, d1.x,d1.y,d1.z,d1.w,
                        d2.x,d2.y,d2.z,d2.w, d3.x,d3.y,d3.z,d3.w};
        BfTile P;
        #pragma unroll
        for (int e = 0; e < 16; ++e) {
            int off = hi * 16 + e;
            float ev = sc + dd[e];
            ev = ev > 0.f ? ev : NEG_SLOPE * ev;
            float pp = ((bits >> off) & 1u) ? __expf(ev - mx) : 0.f;
            P.h[e] = f2bf(pp);
        }

        #pragma unroll
        for (int t = 0; t < 8; ++t) {
            const unsigned short* ap = buf + (t * 16 + l16) * 40 + hi * 16;
            BfTile A;
            A.q[0] = *(const uint4*)(ap);        // cols 16*hi..16*hi+7
            A.q[1] = *(const uint4*)(ap + 8);    // cols 16*hi+8..16*hi+15
            acc[t] = __builtin_amdgcn_wmma_f32_16x16x32_bf16(
                false, A.v, false, P.v, (short)0, acc[t], false, false);
        }
    }

    // epilogue: /= s, ELU, store fp32
    float sv  = srow[irow];
    float inv = sv > 0.f ? 1.0f / sv : 0.f;
    float* hp = Hout + (size_t)irow * DIM + nbase + hi * 8;
    #pragma unroll
    for (int t = 0; t < 8; ++t) {
        float o[8];
        #pragma unroll
        for (int r = 0; r < 8; ++r) {
            float x = acc[t][r] * inv;
            o[r] = x > 0.f ? x : (__expf(x) - 1.0f);
        }
        float4 o0 = {o[0], o[1], o[2], o[3]};
        float4 o1 = {o[4], o[5], o[6], o[7]};
        *(float4*)(hp + t * 16)     = o0;
        *(float4*)(hp + t * 16 + 4) = o1;
    }
}

// ---------------------------------------------------------------------------
extern "C" void kernel_launch(void* const* d_in, const int* in_sizes, int n_in,
                              void* d_out, int out_size, void* d_ws, size_t ws_size,
                              hipStream_t stream) {
    const float* V   = (const float*)d_in[0];
    const int*   adj = (const int*)d_in[1];
    const float* W1  = (const float*)d_in[2];
    const float* as1 = (const float*)d_in[3];
    const float* ad1 = (const float*)d_in[4];
    const float* W2  = (const float*)d_in[5];
    const float* as2 = (const float*)d_in[6];
    const float* ad2 = (const float*)d_in[7];
    float* Hout = (float*)d_out;

    char* ws = (char*)d_ws;
    unsigned int*   packed = (unsigned int*)ws;   ws += (size_t)NNODES * NWORDS * 4;   // 8 MB
    unsigned short* WhT    = (unsigned short*)ws; ws += (size_t)DIM * NNODES * 2;      // 4 MB
    float*          H1     = (float*)ws;          ws += (size_t)NNODES * DIM * 4;      // 8 MB
    unsigned short* W1T    = (unsigned short*)ws; ws += (size_t)DIM * DIM * 2;
    unsigned short* W2T    = (unsigned short*)ws; ws += (size_t)DIM * DIM * 2;
    float* srcv = (float*)ws; ws += (size_t)NNODES * 4;
    float* dstv = (float*)ws; ws += (size_t)NNODES * 4;
    float* mr   = (float*)ws; ws += (size_t)NNODES * 4;
    float* sr   = (float*)ws; ws += (size_t)NNODES * 4;

    k_wt<<<DIM, DIM, 0, stream>>>(W1, W1T);
    k_wt<<<DIM, DIM, 0, stream>>>(W2, W2T);
    k_pack<<<((size_t)NNODES * NNODES) / 256, 256, 0, stream>>>(adj, packed);

    // layer 1
    k_gemm<<<NNODES / 128, 256, 0, stream>>>(V, W1T, as1, ad1, WhT, srcv, dstv);
    k_stats<<<NNODES / 8, 256, 0, stream>>>(packed, srcv, dstv, mr, sr);
    k_attn<<<dim3(NNODES / 128, 2), 256, 0, stream>>>(WhT, packed, srcv, dstv, mr, sr, H1);

    // layer 2
    k_gemm<<<NNODES / 128, 256, 0, stream>>>(H1, W2T, as2, ad2, WhT, srcv, dstv);
    k_stats<<<NNODES / 8, 256, 0, stream>>>(packed, srcv, dstv, mr, sr);
    k_attn<<<dim3(NNODES / 128, 2), 256, 0, stream>>>(WhT, packed, srcv, dstv, mr, sr, Hout);
}
